// Decoder_9070970929218
// MI455X (gfx1250) — matrix operand order repaired and checked by reading, not compiled, not run on hardware
//
#include <hip/hip_runtime.h>
#include <hip/hip_bf16.h>
#include <math.h>

// ---------------------------------------------------------------------------
// Problem dimensions (fixed by the reference)
// ---------------------------------------------------------------------------
#define B_DIM 16
#define T_DIM 512
#define E_DIM 1024
#define D_DIM 1024
#define A_DIM 1024
#define O_DIM 10000
#define L_DIM 128
#define N4D   4096                 // 4*D
#define XDIM  2048                 // D + E  (LSTM0 input)
#define SOS_ID (O_DIM - 1)
#define NSTEP (L_DIM + 1)          // 129 decoder steps
#define NROW  (B_DIM * NSTEP)      // 2064 output rows

typedef __bf16 bf16;
typedef __attribute__((ext_vector_type(16))) bf16  v16bf;
typedef __attribute__((ext_vector_type(8)))  bf16  v8bf;
typedef __attribute__((ext_vector_type(8)))  float v8f;

// ---------------------------------------------------------------------------
// WMMA helpers (CDNA5, wave32). Fragment layouts per ISA cdna5 §7.12.2:
//  A (16x32 bf16):  lane = g*16+m holds row m; VGPR v holds K pair
//                   kbase = v<4 ? 2v+8g : 16+2(v-4)+8g
//  B (32x16 bf16):  lane = g*16+n holds col n; element i holds K = i<8 ? 8g+i : 16+8g+(i-8)
//  C/D (16x16 f32): lane = g*16+n holds col n; VGPR r holds row r+8g
// ---------------------------------------------------------------------------
__device__ __forceinline__ v8f wmma_bf16(v16bf a, v16bf b, v8f c) {
  // (neg_a, A, neg_b, B, c_mod, C, reuse_a, reuse_b)
  return __builtin_amdgcn_wmma_f32_16x16x32_bf16(false, a, false, b,
                                                 (short)0, c, false, false);
}

// Inverse of the A-fragment mapping: element offset within a 512-elem
// (16 lanes x 2 halves x 16 bf16) packed k-tile for matrix element
// (row mi in 0..15, col kk in 0..31).
__device__ __forceinline__ int afrag_off(int mi, int kk) {
  const int g = (kk >> 3) & 1;            // lane half
  const int t = kk & 7;
  const int v = ((kk & 16) ? 4 : 0) + (t >> 1);
  const int elem = 2 * v + (t & 1);
  return (g * 16 + mi) * 16 + elem;
}

// A fragment from fragment-packed storage: P[ktile][lane][16] contiguous.
// One 32-byte contiguous read per lane (lowered to b128 loads).
__device__ __forceinline__ v16bf load_a_frag_packed(const bf16* P, int K,
                                                    int row0, int k0) {
  const int lane = threadIdx.x & 31;
  const long tile = (long)(row0 >> 4) * (K >> 5) + (k0 >> 5);
  const bf16* base = P + tile * 512 + lane * 16;
  v16bf a;
#pragma unroll
  for (int i = 0; i < 16; ++i) a[i] = base[i];
  return a;
}

// B fragment where the logical KxN matrix is stored transposed as W[N, K]
// row-major. Per lane this is two 16-byte reads (K 8g..8g+7, 16+8g..16+8g+7).
__device__ __forceinline__ v16bf load_b_frag_wt(const bf16* W, int ldk,
                                                int n0, int k0) {
  const int lane = threadIdx.x & 31;
  const int n = lane & 15;
  const int g = lane >> 4;
  const bf16* base = W + (long)(n0 + n) * ldk + k0 + g * 8;
  v16bf b;
#pragma unroll
  for (int i = 0; i < 16; ++i) b[i] = base[i < 8 ? i : i + 8];
  return b;
}

// ---------------------------------------------------------------------------
// Conversion / packing kernels (one-time per launch)
// ---------------------------------------------------------------------------
// plain f32 -> bf16 (for [N,K]-stored LSTM weights; already B-friendly)
__global__ void f2bf_kernel(const float* __restrict__ src,
                            bf16* __restrict__ dst, long n) {
  long i = (long)blockIdx.x * blockDim.x + threadIdx.x;
  if (i < n) dst[i] = (bf16)src[i];
}

// f32 [K, N] row-major  ->  bf16 [N, K] row-major (transpose pack)
__global__ void f2bf_t_kernel(const float* __restrict__ src,
                              bf16* __restrict__ dst, int K, int N) {
  long i = (long)blockIdx.x * blockDim.x + threadIdx.x;
  if (i >= (long)K * N) return;
  const int n = (int)(i / K), k = (int)(i % K);
  dst[i] = (bf16)src[(long)k * N + n];
}

// f32 [M, K] row-major -> A-fragment-packed bf16 (M mult of 16, K mult of 32)
__global__ void f2bf_packA_kernel(const float* __restrict__ src,
                                  bf16* __restrict__ dst, int M, int K) {
  long i = (long)blockIdx.x * blockDim.x + threadIdx.x;
  if (i >= (long)M * K) return;
  const int m = (int)(i / K), k = (int)(i % K);
  const long tile = (long)(m >> 4) * (K >> 5) + (k >> 5);
  dst[tile * 512 + afrag_off(m & 15, k & 31)] = (bf16)src[i];
}

__global__ void zero_kernel(unsigned int* __restrict__ p, long nwords) {
  long i = (long)blockIdx.x * blockDim.x + threadIdx.x;
  if (i < nwords) p[i] = 0u;
}

// ---------------------------------------------------------------------------
// pre_enc = bf16(tanh(hpad @ att_Wenc))   M=B*T=8192, N=A=1024, K=E=1024
// A from packed hpad, B from transposed Wenc. block=256 (8 waves), grid (512,8)
// pre_enc is consumed elementwise (attw) -> stored row-major [B*T, A].
// ---------------------------------------------------------------------------
__global__ void preenc_kernel(const bf16* __restrict__ hpad_pack,
                              const bf16* __restrict__ wenc_t,
                              bf16* __restrict__ pre_enc) {
  const int wave = threadIdx.x >> 5;
  const int row0 = blockIdx.x * 16;
  const int n0 = (blockIdx.y * 8 + wave) * 16;
  v8f acc = {};
  for (int k0 = 0; k0 < E_DIM; k0 += 32) {
    v16bf a = load_a_frag_packed(hpad_pack, E_DIM, row0, k0);
    v16bf b = load_b_frag_wt(wenc_t, E_DIM, n0, k0);
    acc = wmma_bf16(a, b, acc);
  }
  const int lane = threadIdx.x & 31;
  const int n = lane & 15, g = lane >> 4;
#pragma unroll
  for (int r = 0; r < 8; ++r) {
    const long row = row0 + r + 8 * g;
    pre_enc[row * A_DIM + n0 + n] = (bf16)tanhf(acc[r]);
  }
}

// ---------------------------------------------------------------------------
// dq = bf16(tanh(z0 @ att_Wdec))   M=16, N=A=1024, K=D=1024
// dq consumed elementwise -> row-major. block=256 (8 waves), grid=8
// ---------------------------------------------------------------------------
__global__ void dq_kernel(const bf16* __restrict__ z0_pack,
                          const bf16* __restrict__ wdec_t,
                          bf16* __restrict__ dq) {
  const int wave = threadIdx.x >> 5;
  const int n0 = (blockIdx.x * 8 + wave) * 16;
  v8f acc = {};
  for (int k0 = 0; k0 < D_DIM; k0 += 32) {
    v16bf a = load_a_frag_packed(z0_pack, D_DIM, 0, k0);
    v16bf b = load_b_frag_wt(wdec_t, D_DIM, n0, k0);
    acc = wmma_bf16(a, b, acc);
  }
  const int lane = threadIdx.x & 31;
  const int n = lane & 15, g = lane >> 4;
#pragma unroll
  for (int r = 0; r < 8; ++r)
    dq[(r + 8 * g) * A_DIM + n0 + n] = (bf16)tanhf(acc[r]);
}

// ---------------------------------------------------------------------------
// escore[b,t] = <pre_enc[b,t,:], dq[b,:]> ; w = softmax(2*escore) per b.
// one block of 512 threads per b (thread t owns time step t); 16-byte loads.
// ---------------------------------------------------------------------------
__global__ void attw_kernel(const bf16* __restrict__ pre_enc,
                            const bf16* __restrict__ dq,
                            float* __restrict__ w) {
  const int b = blockIdx.x;
  const int t = threadIdx.x;
  const bf16* pe = pre_enc + ((long)b * T_DIM + t) * A_DIM;
  const bf16* q  = dq + (long)b * A_DIM;
  float e = 0.f;
  for (int a = 0; a < A_DIM; a += 8) {
    const v8bf pv = *(const v8bf*)(pe + a);
    const v8bf qv = *(const v8bf*)(q + a);
#pragma unroll
    for (int i = 0; i < 8; ++i) e += (float)pv[i] * (float)qv[i];
  }
  e *= 2.0f;  // SCALING
  __shared__ float red[T_DIM];
  red[t] = e;
  __syncthreads();
  for (int s = T_DIM / 2; s > 0; s >>= 1) {
    if (t < s) red[t] = fmaxf(red[t], red[t + s]);
    __syncthreads();
  }
  const float m = red[0];
  __syncthreads();
  const float ex = expf(e - m);
  red[t] = ex;
  __syncthreads();
  for (int s = T_DIM / 2; s > 0; s >>= 1) {
    if (t < s) red[t] += red[t + s];
    __syncthreads();
  }
  w[b * T_DIM + t] = ex / red[0];
}

// ---------------------------------------------------------------------------
// att_c[b,e] = sum_t w[b,t] * hpad[b,t,e] -> x0_pack[b, D+e] (A-frag layout)
// grid = (B), block = 128; each thread owns 8 consecutive e (v8bf loads).
// ---------------------------------------------------------------------------
__global__ void attc_kernel(const float* __restrict__ w,
                            const bf16* __restrict__ hpad,
                            bf16* __restrict__ x0_pack) {
  const int b = blockIdx.x;
  const int e0 = threadIdx.x * 8;  // 128 threads * 8 = 1024
  __shared__ float wl[T_DIM];
  for (int t = threadIdx.x; t < T_DIM; t += blockDim.x)
    wl[t] = w[b * T_DIM + t];
  __syncthreads();
  float acc[8];
#pragma unroll
  for (int i = 0; i < 8; ++i) acc[i] = 0.f;
  const bf16* hb = hpad + (long)b * T_DIM * E_DIM + e0;
  for (int t = 0; t < T_DIM; ++t) {
    const v8bf hv = *(const v8bf*)(hb + (long)t * E_DIM);
    const float wt_ = wl[t];
#pragma unroll
    for (int i = 0; i < 8; ++i) acc[i] += wt_ * (float)hv[i];
  }
#pragma unroll
  for (int i = 0; i < 8; ++i) {
    const int k = D_DIM + e0 + i;  // column in x0 [16, XDIM]
    x0_pack[(long)(k >> 5) * 512 + afrag_off(b, k & 31)] = (bf16)acc[i];
  }
}

// ---------------------------------------------------------------------------
// ey = embed_w[ys_in[b, step]] -> x0_pack[b, 0:D] (A-frag layout)
// grid = (B, D/256), block = 256
// ---------------------------------------------------------------------------
__global__ void embed_kernel(const int* __restrict__ ys,
                             const float* __restrict__ embed_w,
                             bf16* __restrict__ x0_pack, int step) {
  const int b = blockIdx.x;
  const int j = blockIdx.y * blockDim.x + threadIdx.x;
  const int tok = (step == 0) ? SOS_ID : ys[b * L_DIM + (step - 1)];
  x0_pack[(long)(j >> 5) * 512 + afrag_off(b, j & 31)] =
      (bf16)embed_w[(long)tok * D_DIM + j];
}

// ---------------------------------------------------------------------------
// LSTM gate GEMM: g[16, 4096] = X[16,kx] @ Wih^T + H[16,kh] @ Whh^T
// X/H fragment-packed; Wih/Whh: [4096, k] row-major bf16.
// block = 256 (8 waves), grid = 32 (256 N-tiles)
// ---------------------------------------------------------------------------
__global__ void gates_kernel(const bf16* __restrict__ Xp, int kx,
                             const bf16* __restrict__ Wih,
                             const bf16* __restrict__ Hp, int kh,
                             const bf16* __restrict__ Whh,
                             float* __restrict__ g) {
  const int wave = threadIdx.x >> 5;
  const int n0 = (blockIdx.x * 8 + wave) * 16;
  // warm L2 for this wave's weight stripes (global_prefetch_b8)
  __builtin_prefetch(Wih + (long)n0 * kx, 0, 1);
  __builtin_prefetch(Whh + (long)n0 * kh, 0, 1);
  v8f acc = {};
  for (int k0 = 0; k0 < kx; k0 += 32) {
    v16bf a = load_a_frag_packed(Xp, kx, 0, k0);
    v16bf b = load_b_frag_wt(Wih, kx, n0, k0);
    acc = wmma_bf16(a, b, acc);
  }
  for (int k0 = 0; k0 < kh; k0 += 32) {
    v16bf a = load_a_frag_packed(Hp, kh, 0, k0);
    v16bf b = load_b_frag_wt(Whh, kh, n0, k0);
    acc = wmma_bf16(a, b, acc);
  }
  const int lane = threadIdx.x & 31;
  const int n = lane & 15, gq = lane >> 4;
#pragma unroll
  for (int r = 0; r < 8; ++r)
    g[(long)(r + 8 * gq) * N4D + n0 + n] = acc[r];
}

// ---------------------------------------------------------------------------
// LSTM cell elementwise: gate order [i, f, g, o]; updates c (f32) and writes
// h (bf16) in A-fragment-packed layout. One thread per (b, j).
// ---------------------------------------------------------------------------
__global__ void cell_kernel(const float* __restrict__ g,
                            const float* __restrict__ b_ih,
                            const float* __restrict__ b_hh,
                            float* __restrict__ c,
                            bf16* __restrict__ h_pack) {
  const int idx = blockIdx.x * blockDim.x + threadIdx.x;  // 0..B*D-1
  const int b = idx >> 10, j = idx & (D_DIM - 1);
  const float* gb = g + (long)b * N4D;
  const float gi = gb[j]             + b_ih[j]             + b_hh[j];
  const float gf = gb[D_DIM + j]     + b_ih[D_DIM + j]     + b_hh[D_DIM + j];
  const float gg = gb[2 * D_DIM + j] + b_ih[2 * D_DIM + j] + b_hh[2 * D_DIM + j];
  const float go = gb[3 * D_DIM + j] + b_ih[3 * D_DIM + j] + b_hh[3 * D_DIM + j];
  const float si = 1.f / (1.f + expf(-gi));
  const float sf = 1.f / (1.f + expf(-gf));
  const float so = 1.f / (1.f + expf(-go));
  const float cn = sf * c[idx] + si * tanhf(gg);
  c[idx] = cn;
  h_pack[(long)(j >> 5) * 512 + afrag_off(b, j & 31)] = (bf16)(so * tanhf(cn));
}

// ---------------------------------------------------------------------------
// Output projection fused with online log-softmax / NLL / argmax.
// One block per decoder step t (129 blocks): rows = all 16 batch entries of
// z_all packed slot t+1. 8 waves stream 128-col chunks via WMMA into LDS;
// threads 0..15 keep per-row running (max, sumexp, label logit, argmax).
// ---------------------------------------------------------------------------
__global__ void out_loss_kernel(const bf16* __restrict__ z_all_pack,
                                const bf16* __restrict__ wout_t,  // [O, D]
                                const float* __restrict__ b_out,
                                const int* __restrict__ ys,
                                float* __restrict__ nll,
                                float* __restrict__ correct) {
  const int t = blockIdx.x;  // 0..128
  const bf16* Ap = z_all_pack + (long)(t + 1) * B_DIM * D_DIM;  // packed slot
  __shared__ float tile[B_DIM * 128];
  const int wave = threadIdx.x >> 5;
  const int lane = threadIdx.x & 31;
  const int myrow = threadIdx.x;  // row b for tid < 16

  float m = -INFINITY, ssum = 0.f, labv = 0.f, best = -INFINITY;
  int bestidx = 0, label = 0;
  if (threadIdx.x < B_DIM)
    label = (t < L_DIM) ? ys[myrow * L_DIM + t] : (O_DIM - 1);

  const int nchunk = (O_DIM + 127) / 128;  // 79 (O_DIM % 16 == 0)
  for (int chunk = 0; chunk < nchunk; ++chunk) {
    const int n0 = chunk * 128 + wave * 16;
    v8f acc = {};
    if (n0 < O_DIM) {  // wave-uniform predicate; EXEC all-1 inside
      for (int k0 = 0; k0 < D_DIM; k0 += 32) {
        v16bf a = load_a_frag_packed(Ap, D_DIM, 0, k0);
        v16bf b = load_b_frag_wt(wout_t, D_DIM, n0, k0);
        acc = wmma_bf16(a, b, acc);
      }
    }
    __syncthreads();  // previous chunk's consumers done with `tile`
    {
      const int n = lane & 15, g = lane >> 4;
#pragma unroll
      for (int r = 0; r < 8; ++r) {
        const float v = (n0 < O_DIM) ? (acc[r] + b_out[n0 + n]) : -INFINITY;
        tile[(r + 8 * g) * 128 + wave * 16 + n] = v;
      }
    }
    __syncthreads();
    if (threadIdx.x < B_DIM) {
      for (int cc = 0; cc < 128; ++cc) {
        const int col = chunk * 128 + cc;
        if (col >= O_DIM) break;
        const float v = tile[myrow * 128 + cc];
        const float mn = fmaxf(m, v);
        ssum = ssum * expf(m - mn) + expf(v - mn);
        m = mn;
        if (col == label) labv = v;
        if (v > best) { best = v; bestidx = col; }
      }
    }
  }
  if (threadIdx.x < B_DIM) {
    const int row = t * B_DIM + myrow;          // ordering irrelevant to mean
    nll[row] = (m + logf(ssum)) - labv;         // -log_softmax[label]
    correct[row] = (bestidx == label) ? 1.f : 0.f;
  }
}

// loss = mean(nll) * L ; acc = mean(correct)
__global__ void loss_reduce_kernel(const float* __restrict__ nll,
                                   const float* __restrict__ correct,
                                   float* __restrict__ out) {
  __shared__ float s1[256], s2[256];
  float a = 0.f, b = 0.f;
  for (int i = threadIdx.x; i < NROW; i += 256) { a += nll[i]; b += correct[i]; }
  s1[threadIdx.x] = a; s2[threadIdx.x] = b;
  __syncthreads();
  for (int s = 128; s > 0; s >>= 1) {
    if (threadIdx.x < s) {
      s1[threadIdx.x] += s1[threadIdx.x + s];
      s2[threadIdx.x] += s2[threadIdx.x + s];
    }
    __syncthreads();
  }
  if (threadIdx.x == 0) {
    out[0] = s1[0] / (float)NROW * (float)L_DIM;
    out[1] = s2[0] / (float)NROW;
  }
}

// ---------------------------------------------------------------------------
// Host side
// ---------------------------------------------------------------------------
static inline long cdiv(long a, long b) { return (a + b - 1) / b; }

extern "C" void kernel_launch(void* const* d_in, const int* in_sizes, int n_in,
                              void* d_out, int out_size, void* d_ws, size_t ws_size,
                              hipStream_t stream) {
  (void)in_sizes; (void)n_in; (void)out_size; (void)ws_size;
  const float* hpad_f   = (const float*)d_in[0];
  // d_in[1] = hlen (all == T, mask is identity) -- unused
  const int*   ys       = (const int*)d_in[2];
  const float* embed_w  = (const float*)d_in[3];
  const float* wenc_f   = (const float*)d_in[4];
  const float* wdec_f   = (const float*)d_in[5];
  const float* wih0_f   = (const float*)d_in[6];
  const float* whh0_f   = (const float*)d_in[7];
  const float* b_ih0    = (const float*)d_in[8];
  const float* b_hh0    = (const float*)d_in[9];
  const float* wih1_f   = (const float*)d_in[10];
  const float* whh1_f   = (const float*)d_in[11];
  const float* b_ih1    = (const float*)d_in[12];
  const float* b_hh1    = (const float*)d_in[13];
  const float* wout_f   = (const float*)d_in[14];
  const float* b_out    = (const float*)d_in[15];
  float* out = (float*)d_out;

  // ---- workspace layout (bf16 weight cache + per-step buffers) ----
  char* ws = (char*)d_ws;
  size_t off = 0;
  auto alloc = [&](size_t bytes) -> char* {
    char* p = ws + off;
    off += (bytes + 255) & ~(size_t)255;
    return p;
  };
  bf16* hpad_bf    = (bf16*)alloc((size_t)B_DIM * T_DIM * E_DIM * 2);  // row-major (attc)
  bf16* hpad_pack  = (bf16*)alloc((size_t)B_DIM * T_DIM * E_DIM * 2);  // A-frag (preenc)
  bf16* preenc_bf  = (bf16*)alloc((size_t)B_DIM * T_DIM * A_DIM * 2);
  bf16* wenc_t     = (bf16*)alloc((size_t)E_DIM * A_DIM * 2);          // [A, E]
  bf16* wdec_t     = (bf16*)alloc((size_t)D_DIM * A_DIM * 2);          // [A, D]
  bf16* wih0_bf    = (bf16*)alloc((size_t)N4D * XDIM * 2);             // [4D, D+E]
  bf16* whh0_bf    = (bf16*)alloc((size_t)N4D * D_DIM * 2);
  bf16* wih1_bf    = (bf16*)alloc((size_t)N4D * D_DIM * 2);
  bf16* whh1_bf    = (bf16*)alloc((size_t)N4D * D_DIM * 2);
  bf16* wout_t     = (bf16*)alloc((size_t)D_DIM * O_DIM * 2);          // [O, D]
  bf16* z_all_pack = (bf16*)alloc((size_t)(NSTEP + 1) * B_DIM * D_DIM * 2);
  bf16* z0_pack    = (bf16*)alloc((size_t)B_DIM * D_DIM * 2);
  bf16* dq_bf      = (bf16*)alloc((size_t)B_DIM * A_DIM * 2);
  bf16* x0_pack    = (bf16*)alloc((size_t)B_DIM * XDIM * 2);
  float* c0        = (float*)alloc((size_t)B_DIM * D_DIM * 4);
  float* c1        = (float*)alloc((size_t)B_DIM * D_DIM * 4);
  float* w_att     = (float*)alloc((size_t)B_DIM * T_DIM * 4);
  float* g0        = (float*)alloc((size_t)B_DIM * N4D * 4);
  float* g1        = (float*)alloc((size_t)B_DIM * N4D * 4);
  float* nll       = (float*)alloc((size_t)NROW * 4);
  float* correct   = (float*)alloc((size_t)NROW * 4);

  // ---- one-time conversions / packings (weights stay L2-resident) ----
  auto cvt = [&](const float* s, bf16* d, long n) {
    f2bf_kernel<<<dim3((unsigned)cdiv(n, 256)), dim3(256), 0, stream>>>(s, d, n);
  };
  auto cvtT = [&](const float* s, bf16* d, int K, int N) {
    long n = (long)K * N;
    f2bf_t_kernel<<<dim3((unsigned)cdiv(n, 256)), dim3(256), 0, stream>>>(s, d, K, N);
  };
  auto packA = [&](const float* s, bf16* d, int M, int K) {
    long n = (long)M * K;
    f2bf_packA_kernel<<<dim3((unsigned)cdiv(n, 256)), dim3(256), 0, stream>>>(s, d, M, K);
  };
  cvt(hpad_f, hpad_bf, (long)B_DIM * T_DIM * E_DIM);
  packA(hpad_f, hpad_pack, B_DIM * T_DIM, E_DIM);
  cvtT(wenc_f, wenc_t, E_DIM, A_DIM);
  cvtT(wdec_f, wdec_t, D_DIM, A_DIM);
  cvt(wih0_f, wih0_bf, (long)N4D * XDIM);
  cvt(whh0_f, whh0_bf, (long)N4D * D_DIM);
  cvt(wih1_f, wih1_bf, (long)N4D * D_DIM);
  cvt(whh1_f, whh1_bf, (long)N4D * D_DIM);
  cvtT(wout_f, wout_t, D_DIM, O_DIM);

  // ---- zero initial recurrent state ----
  auto zero = [&](void* p, long bytes) {
    long nw = bytes / 4;
    zero_kernel<<<dim3((unsigned)cdiv(nw, 256)), dim3(256), 0, stream>>>(
        (unsigned int*)p, nw);
  };
  zero(z0_pack, (long)B_DIM * D_DIM * 2);
  zero(z_all_pack, (long)B_DIM * D_DIM * 2);  // slot 0 == z1 at step -1
  zero(c0, (long)B_DIM * D_DIM * 4);
  zero(c1, (long)B_DIM * D_DIM * 4);

  // ---- pre_enc = tanh(hpad @ Wenc)  (WMMA bf16) ----
  preenc_kernel<<<dim3(B_DIM * T_DIM / 16, A_DIM / (16 * 8)), dim3(256), 0,
                  stream>>>(hpad_pack, wenc_t, preenc_bf);

  // ---- 129 sequential decoder steps ----
  for (int step = 0; step < NSTEP; ++step) {
    bf16* z1_prev = z_all_pack + (size_t)step * B_DIM * D_DIM;
    bf16* z1_cur  = z_all_pack + (size_t)(step + 1) * B_DIM * D_DIM;

    embed_kernel<<<dim3(B_DIM, D_DIM / 256), dim3(256), 0, stream>>>(
        ys, embed_w, x0_pack, step);
    dq_kernel<<<dim3(A_DIM / (16 * 8)), dim3(256), 0, stream>>>(
        z0_pack, wdec_t, dq_bf);
    attw_kernel<<<dim3(B_DIM), dim3(T_DIM), 0, stream>>>(
        preenc_bf, dq_bf, w_att);
    attc_kernel<<<dim3(B_DIM), dim3(128), 0, stream>>>(
        w_att, hpad_bf, x0_pack);
    gates_kernel<<<dim3(N4D / (16 * 8)), dim3(256), 0, stream>>>(
        x0_pack, XDIM, wih0_bf, z0_pack, D_DIM, whh0_bf, g0);
    cell_kernel<<<dim3(B_DIM * D_DIM / 256), dim3(256), 0, stream>>>(
        g0, b_ih0, b_hh0, c0, z0_pack);
    gates_kernel<<<dim3(N4D / (16 * 8)), dim3(256), 0, stream>>>(
        z0_pack, D_DIM, wih1_bf, z1_prev, D_DIM, whh1_bf, g1);
    cell_kernel<<<dim3(B_DIM * D_DIM / 256), dim3(256), 0, stream>>>(
        g1, b_ih1, b_hh1, c1, z1_cur);
  }

  // ---- fused output GEMM + online log-softmax / NLL / argmax ----
  out_loss_kernel<<<dim3(NSTEP), dim3(256), 0, stream>>>(
      z_all_pack, wout_t, b_out, ys, nll, correct);
  loss_reduce_kernel<<<dim3(1), dim3(256), 0, stream>>>(nll, correct, out);
}
